// OPTLayer_67860483277520
// MI455X (gfx1250) — matrix operand order repaired and checked by reading, not compiled, not run on hardware
//
#include <hip/hip_runtime.h>
#include <hip/hip_bf16.h>

// ---------------------------------------------------------------------------
// OPT layer for MI455X (gfx1250): bf16 WMMA everywhere, f32 accumulation.
//   x:[2,2048,1024] f32, weights f32 [out,in]; y = x @ W^T chains + attention.
// Matmuls -> v_wmma_f32_16x16x32_bf16 (wave32). GEMM A-panels are fed by the
// Tensor Data Mover (tensor_load_to_lds, TENSORcnt) when the builtin exists.
// ---------------------------------------------------------------------------

typedef __attribute__((ext_vector_type(16))) __bf16 v16bf;
typedef __attribute__((ext_vector_type(8)))  __bf16 v8bf;
typedef __attribute__((ext_vector_type(8)))  float  v8f;
typedef __attribute__((ext_vector_type(4)))  unsigned int u32x4;
typedef __attribute__((ext_vector_type(8)))  int i32x8;
typedef __attribute__((ext_vector_type(4)))  int i32x4;
typedef __attribute__((address_space(3)))    __bf16 lds_bf16;

#define DIM      1024
#define FFN_DIM  4096
#define SEQ      2048
#define BATCH    2
#define NHEADS   16
#define HDIM     64
#define ROWS     (BATCH * SEQ)   // 4096 token rows

#if __has_builtin(__builtin_amdgcn_tensor_load_to_lds) && \
    __has_builtin(__builtin_amdgcn_s_wait_tensorcnt)
#define USE_TDM 1
#else
#define USE_TDM 0
#endif

// ---------------------------------------------------------------------------
// f32 -> bf16 conversion (round-to-nearest-even via HW cvt)
// ---------------------------------------------------------------------------
__global__ void cvt_f32_bf16(const float* __restrict__ in,
                             __bf16* __restrict__ out, int n) {
    int i = (blockIdx.x * blockDim.x + threadIdx.x) * 4;
    if (i + 3 < n) {
        float4 v = *(const float4*)(in + i);
        out[i + 0] = (__bf16)v.x;
        out[i + 1] = (__bf16)v.y;
        out[i + 2] = (__bf16)v.z;
        out[i + 3] = (__bf16)v.w;
    } else {
        for (; i < n; ++i) out[i] = (__bf16)in[i];
    }
}

// ---------------------------------------------------------------------------
// GEMM: C[M,N] = A[M,K] (bf16, row-major) * W[N,K]^T (bf16, row-major)
// Workgroup: 256 threads = 8 waves; tile 128M x 64N; K-step 32.
//  - A panel (128 x 32) DMA'd into double-buffered LDS by the TDM (wave 0
//    issues; TENSORcnt ordering; DMA of step s+1 overlaps WMMA of step s).
//  - W tile staged into LDS (table order) by all threads; B-frag = lane (k-half, n), 16
//    k values (aligned 32B LDS reads).
// ---------------------------------------------------------------------------
template <typename OutT>
__global__ __launch_bounds__(256)
void gemm_bf16_wmma(const __bf16* __restrict__ A,
                    const __bf16* __restrict__ W,
                    OutT* __restrict__ C,
                    int M, int N, int K) {
    __shared__ __bf16 ldsWt[32][64];        // [h*16+n%16][(n/16)*16+s] W tile, 4 KB
#if USE_TDM
    __shared__ __bf16 ldsA[2][128][32];     // double-buffered A panel, 16 KB
#endif

    const int tid  = threadIdx.x;
    const int wave = tid >> 5;
    const int lane = tid & 31;
    const int lrow = lane & 15;       // row-in-tile index
    const int half = lane >> 4;       // K-half selector

    const int m0 = blockIdx.y * 128 + wave * 16;
    const int n0 = blockIdx.x * 64;

    v8f acc0 = {}, acc1 = {}, acc2 = {}, acc3 = {};

    // W-staging assignment: each thread moves 8 contiguous K values of one row
    const int sn = tid >> 2;          // 0..63
    const int sk = (tid & 3) * 8;     // 0,8,16,24

    const int nsteps = K >> 5;

#if USE_TDM
    // ---- Tensor DMA descriptor pieces (wave-uniform, SGPR-resident) ----
    const unsigned long long gAbase =
        (unsigned long long)(const void*)(A + (size_t)(blockIdx.y * 128) * K);
    const unsigned ldsOff = (unsigned)(unsigned long long)(lds_bf16*)&ldsA[0][0][0];

    auto issue_tdm = [&](int k0, int buf) {
        unsigned long long ga = gAbase + (unsigned long long)k0 * 2ull;
        // D# group0: count=1 | lds_addr | global_addr[56:0] | type=2
        u32x4 g0 = { 1u,
                     ldsOff + (unsigned)buf * (128u * 32u * 2u),
                     (unsigned)ga,
                     (unsigned)((ga >> 32) & 0x01FFFFFFull) | 0x80000000u };
        // D# group1: data_size=2B | tensor_dim0=K | tensor_dim1=M |
        //            tile_dim0=32 | tile_dim1=128 | tensor_dim0_stride=K
        i32x8 g1 = { (int)0x00010000,
                     (int)(((unsigned)K & 0xFFFFu) << 16),
                     (int)((((unsigned)M & 0xFFFFu) << 16) | ((unsigned)K >> 16)),
                     (int)((32u << 16) | ((unsigned)M >> 16)),
                     128, K, 0, 0 };
        i32x4 z4 = { 0, 0, 0, 0 };
#if __clang_major__ >= 23
        i32x8 z8 = { 0, 0, 0, 0, 0, 0, 0, 0 };
        __builtin_amdgcn_tensor_load_to_lds(g0, g1, z4, z4, z8, 0);
#else
        __builtin_amdgcn_tensor_load_to_lds(g0, g1, z4, z4, 0);
#endif
    };
    if (wave == 0) issue_tdm(0, 0);          // prologue DMA (uniform branch)
#else
    const __bf16* pArow = A + (size_t)(m0 + lrow) * K + half * 8;
#endif

    for (int s = 0; s < nsteps; ++s) {
        const int k0 = s << 5;

        // ---- stage W[n0+sn][k0+sk..+7] into ldsWt[h*16+n%16][(n/16)*16+s] (B-frag table order) ----
        v8bf wv = *(const v8bf*)(W + (size_t)(n0 + sn) * K + k0 + sk);
#pragma unroll
        for (int i = 0; i < 8; ++i) ldsWt[((sk >> 3) & 1) * 16 + (sn & 15)][(sn >> 4) * 16 + (sk >> 4) * 8 + i] = wv[i];

#if USE_TDM
        if (wave == 0) __builtin_amdgcn_s_wait_tensorcnt(0); // A(s) landed
#endif
        __syncthreads();   // W tile + A panel visible; prior reads finished
#if USE_TDM
        if (wave == 0 && s + 1 < nsteps)      // overlap DMA(s+1) with WMMA(s)
            issue_tdm((s + 1) << 5, (s + 1) & 1);

        // ---- A fragment from LDS (row m, K halves {h*8..+7, 16+h*8..+7}) --
        v8bf lo = *(const v8bf*)&ldsA[s & 1][wave * 16 + lrow][half * 8];
        v8bf hi = *(const v8bf*)&ldsA[s & 1][wave * 16 + lrow][half * 8 + 16];
#else
        v8bf lo = *(const v8bf*)(pArow + k0);
        v8bf hi = *(const v8bf*)(pArow + k0 + 16);
        if (k0 + 32 < K) __builtin_prefetch(pArow + k0 + 32, 0, 0);
#endif
        v16bf a;
#pragma unroll
        for (int i = 0; i < 8; ++i) { a[i] = lo[i]; a[8 + i] = hi[i]; }

        // ---- 4 B fragments from LDS: lane = (k-half, n), 16 k in table order ----
        v16bf b0 = *(const v16bf*)&ldsWt[lane][0];
        v16bf b1 = *(const v16bf*)&ldsWt[lane][16];
        v16bf b2 = *(const v16bf*)&ldsWt[lane][32];
        v16bf b3 = *(const v16bf*)&ldsWt[lane][48];

        acc0 = __builtin_amdgcn_wmma_f32_16x16x32_bf16(false, a, false, b0,
                                                       (short)0, acc0, false, false);
        acc1 = __builtin_amdgcn_wmma_f32_16x16x32_bf16(false, a, false, b1,
                                                       (short)0, acc1, false, false);
        acc2 = __builtin_amdgcn_wmma_f32_16x16x32_bf16(false, a, false, b2,
                                                       (short)0, acc2, false, false);
        acc3 = __builtin_amdgcn_wmma_f32_16x16x32_bf16(false, a, false, b3,
                                                       (short)0, acc3, false, false);
        __syncthreads();
    }

    // ---- store: VGPR r -> row m0 + r + half*8, col n0 + j*16 + lrow ----
    const int mrow = m0 + half * 8;
#pragma unroll
    for (int r = 0; r < 8; ++r) {
        size_t base = (size_t)(mrow + r) * N + n0 + lrow;
        C[base +  0] = (OutT)acc0[r];
        C[base + 16] = (OutT)acc1[r];
        C[base + 32] = (OutT)acc2[r];
        C[base + 48] = (OutT)acc3[r];
    }
}

// ---------------------------------------------------------------------------
// Attention: one workgroup per (batch, head, 16-query tile).
//   S = Q K^T  (WMMA, K staged in B-frag table order in per-wave LDS)
//   P = softmax(S) in-place in LDS (f32, exact, no scale per reference)
//   O = P V    (WMMA; V row-major matches B-frag layout directly)
// Dynamic LDS: scores 16x2048 f32 | per-wave K^T tiles | reduce | out accum
// ---------------------------------------------------------------------------
#define S_BYTES   (16 * SEQ * 4)          // 131072
#define KT_BYTES  (8 * HDIM * 16 * 2)     // 16384
#define RED_BYTES (256 * 4)               // 1024
#define OUT_BYTES (16 * HDIM * 4)         // 4096
#define ATTN_SMEM (S_BYTES + KT_BYTES + RED_BYTES + OUT_BYTES)

__global__ __launch_bounds__(256)
void attn_bf16_wmma(const __bf16* __restrict__ Q,
                    const __bf16* __restrict__ Kc,
                    const __bf16* __restrict__ V,
                    __bf16* __restrict__ O) {
    extern __shared__ char smem[];
    float*  S      = (float*)smem;                               // [16][2048]
    __bf16* ldsKt  = (__bf16*)(smem + S_BYTES);                  // 8 x [64][16]
    float*  red    = (float*)(smem + S_BYTES + KT_BYTES);        // [16][16]
    float*  outacc = (float*)(smem + S_BYTES + KT_BYTES + RED_BYTES); // [16][64]
    __shared__ float rowmax[16];
    __shared__ float rowsum[16];

    const int tid  = threadIdx.x;
    const int wave = tid >> 5;
    const int lane = tid & 31;
    const int lrow = lane & 15;
    const int half = lane >> 4;

    const int b  = blockIdx.z;
    const int h  = blockIdx.y;
    const int qt = blockIdx.x;
    const int rowbase = b * SEQ + qt * 16;   // token-row of this query tile
    const int hoff    = h * HDIM;

    for (int i = tid; i < 16 * HDIM; i += 256) outacc[i] = 0.0f;

    // ---- Q fragments for d0 = 0 and 32 (loaded once) ----
    v16bf aq0, aq1;
    {
        const __bf16* pq = Q + (size_t)(rowbase + lrow) * DIM + hoff + half * 8;
#pragma unroll
        for (int i = 0; i < 8; ++i) {
            aq0[i] = pq[i];       aq0[8 + i] = pq[16 + i];
            aq1[i] = pq[32 + i];  aq1[8 + i] = pq[48 + i];
        }
    }

    // ---- scores: S[16, 2048], key tiles strided across the 8 waves ----
    __bf16* kw = ldsKt + wave * HDIM * 16;        // [d/32][d bit 3][kk][16 d, table order] wave-private
    for (int kt = wave; kt < SEQ / 16; kt += 8) {
        // stage K^T: lane loads 32 contiguous d of key row (kt*16 + lrow)
        const __bf16* pk =
            Kc + (size_t)(b * SEQ + kt * 16 + lrow) * DIM + hoff + half * 32;
#pragma unroll
        for (int d = 0; d < 32; ++d) kw[(half * 32 + ((d >> 3) & 1) * 16 + lrow) * 16 + (d >> 4) * 8 + (d & 7)] = pk[d];
        // ds writes/reads from same wave are in-order: no barrier needed

        v16bf b0 = *(const v16bf*)&kw[(lane +  0) * 16];  // d-rows 0..31
        v16bf b1 = *(const v16bf*)&kw[(lane + 32) * 16];  // d-rows 32..63
        v8f c = {};
        c = __builtin_amdgcn_wmma_f32_16x16x32_bf16(false, aq0, false, b0,
                                                    (short)0, c, false, false);
        c = __builtin_amdgcn_wmma_f32_16x16x32_bf16(false, aq1, false, b1,
                                                    (short)0, c, false, false);
#pragma unroll
        for (int r = 0; r < 8; ++r)
            S[(r + half * 8) * SEQ + kt * 16 + lrow] = c[r];
    }
    __syncthreads();

    // ---- softmax over each of the 16 rows (exact, no scaling) ----
    const int sr = tid >> 4;      // row 0..15
    const int sc = tid & 15;      // lane-within-row 0..15
    float mx = -3.4e38f;
    for (int c = sc; c < SEQ; c += 16) mx = fmaxf(mx, S[sr * SEQ + c]);
    red[sr * 16 + sc] = mx;
    __syncthreads();
    if (tid < 16) {
        float m = red[tid * 16];
#pragma unroll
        for (int j = 1; j < 16; ++j) m = fmaxf(m, red[tid * 16 + j]);
        rowmax[tid] = m;
    }
    __syncthreads();
    float sum = 0.0f;
    const float rmx = rowmax[sr];
    for (int c = sc; c < SEQ; c += 16) sum += __expf(S[sr * SEQ + c] - rmx);
    red[sr * 16 + sc] = sum;
    __syncthreads();
    if (tid < 16) {
        float s = red[tid * 16];
#pragma unroll
        for (int j = 1; j < 16; ++j) s += red[tid * 16 + j];
        rowsum[tid] = s;
    }
    __syncthreads();
    const float rinv = 1.0f / rowsum[sr];
    for (int c = sc; c < SEQ; c += 16)
        S[sr * SEQ + c] = __expf(S[sr * SEQ + c] - rmx) * rinv;
    __syncthreads();

    // ---- O = P V : K-dim 2048 in chunks of 32, split across waves ----
    v8f o0 = {}, o1 = {}, o2 = {}, o3 = {};
    for (int kt = wave; kt < SEQ / 32; kt += 8) {
        // A fragment from probabilities (f32 LDS -> bf16)
        const float* ps = S + (size_t)lrow * SEQ + kt * 32 + half * 8;
        v16bf pa;
#pragma unroll
        for (int i = 0; i < 8; ++i) {
            pa[i]     = (__bf16)ps[i];
            pa[8 + i] = (__bf16)ps[16 + i];
        }
        // B fragments from V: lane = d column, 16 key-rows in table order
        const __bf16* pv = V + (size_t)(b * SEQ + kt * 32 + half * 8) * DIM + hoff + lrow;
        v16bf b0, b1, b2, b3;
#pragma unroll
        for (int i = 0; i < 8; ++i) {
            b0[i] = pv[(size_t)i * DIM +  0]; b0[8 + i] = pv[(size_t)(16 + i) * DIM +  0];
            b1[i] = pv[(size_t)i * DIM + 16]; b1[8 + i] = pv[(size_t)(16 + i) * DIM + 16];
            b2[i] = pv[(size_t)i * DIM + 32]; b2[8 + i] = pv[(size_t)(16 + i) * DIM + 32];
            b3[i] = pv[(size_t)i * DIM + 48]; b3[8 + i] = pv[(size_t)(16 + i) * DIM + 48];
        }
        o0 = __builtin_amdgcn_wmma_f32_16x16x32_bf16(false, pa, false, b0,
                                                     (short)0, o0, false, false);
        o1 = __builtin_amdgcn_wmma_f32_16x16x32_bf16(false, pa, false, b1,
                                                     (short)0, o1, false, false);
        o2 = __builtin_amdgcn_wmma_f32_16x16x32_bf16(false, pa, false, b2,
                                                     (short)0, o2, false, false);
        o3 = __builtin_amdgcn_wmma_f32_16x16x32_bf16(false, pa, false, b3,
                                                     (short)0, o3, false, false);
    }
    // combine wave partials in LDS (ds_add_f32 atomics)
#pragma unroll
    for (int r = 0; r < 8; ++r) {
        int orow = (r + half * 8) * HDIM + lrow;
        atomicAdd(&outacc[orow +  0], o0[r]);
        atomicAdd(&outacc[orow + 16], o1[r]);
        atomicAdd(&outacc[orow + 32], o2[r]);
        atomicAdd(&outacc[orow + 48], o3[r]);
    }
    __syncthreads();

    for (int i = tid; i < 16 * HDIM; i += 256) {
        int m = i / HDIM, d = i % HDIM;
        O[(size_t)(rowbase + m) * DIM + hoff + d] = (__bf16)outacc[i];
    }
}

// ---------------------------------------------------------------------------
// Host-side orchestration (graph-capture safe: only launches on `stream`)
// ---------------------------------------------------------------------------
extern "C" void kernel_launch(void* const* d_in, const int* in_sizes, int n_in,
                              void* d_out, int out_size, void* d_ws, size_t ws_size,
                              hipStream_t stream) {
    const float* x  = (const float*)d_in[0];
    const float* wq = (const float*)d_in[1];
    const float* wk = (const float*)d_in[2];
    const float* wv = (const float*)d_in[3];
    const float* wo = (const float*)d_in[4];
    const float* w1 = (const float*)d_in[5];
    const float* w2 = (const float*)d_in[6];
    float* out = (float*)d_out;

    // workspace carve-out (bf16 elements)
    __bf16* ws = (__bf16*)d_ws;
    size_t off = 0;
    auto take = [&](size_t n) { __bf16* p = ws + off; off += n; return p; };
    __bf16* xb  = take((size_t)ROWS * DIM);
    __bf16* wqb = take((size_t)DIM * DIM);
    __bf16* wkb = take((size_t)DIM * DIM);
    __bf16* wvb = take((size_t)DIM * DIM);
    __bf16* wob = take((size_t)DIM * DIM);
    __bf16* w1b = take((size_t)FFN_DIM * DIM);
    __bf16* w2b = take((size_t)DIM * FFN_DIM);
    __bf16* qb  = take((size_t)ROWS * DIM);
    __bf16* kb  = take((size_t)ROWS * DIM);
    __bf16* vb  = take((size_t)ROWS * DIM);
    __bf16* atb = take((size_t)ROWS * DIM);
    __bf16* xob = take((size_t)ROWS * DIM);
    __bf16* x1b = take((size_t)ROWS * FFN_DIM);

    auto cvt = [&](const float* src, __bf16* dst, int n) {
        int blocks = (n / 4 + 255) / 256;
        cvt_f32_bf16<<<blocks, 256, 0, stream>>>(src, dst, n);
    };
    cvt(x,  xb,  ROWS * DIM);
    cvt(wq, wqb, DIM * DIM);
    cvt(wk, wkb, DIM * DIM);
    cvt(wv, wvb, DIM * DIM);
    cvt(wo, wob, DIM * DIM);
    cvt(w1, w1b, FFN_DIM * DIM);
    cvt(w2, w2b, DIM * FFN_DIM);

    dim3 gP(DIM / 64, ROWS / 128);       // 1024-wide GEMMs
    dim3 gF(FFN_DIM / 64, ROWS / 128);   // 4096-wide GEMM (FFN1)

    // Q/K/V projections
    gemm_bf16_wmma<__bf16><<<gP, 256, 0, stream>>>(xb, wqb, qb, ROWS, DIM, DIM);
    gemm_bf16_wmma<__bf16><<<gP, 256, 0, stream>>>(xb, wkb, kb, ROWS, DIM, DIM);
    gemm_bf16_wmma<__bf16><<<gP, 256, 0, stream>>>(xb, wvb, vb, ROWS, DIM, DIM);

    // attention (per batch/head/16-query tile)
    attn_bf16_wmma<<<dim3(SEQ / 16, NHEADS, BATCH), 256, ATTN_SMEM, stream>>>(
        qb, kb, vb, atb);

    // output projection + FFN
    gemm_bf16_wmma<__bf16><<<gP, 256, 0, stream>>>(atb, wob, xob, ROWS, DIM, DIM);
    gemm_bf16_wmma<__bf16><<<gF, 256, 0, stream>>>(xob, w1b, x1b, ROWS, FFN_DIM, DIM);
    gemm_bf16_wmma<float ><<<gP, 256, 0, stream>>>(x1b, w2b, out, ROWS, DIM, FFN_DIM);
}
